// Attention_16054587753058
// MI455X (gfx1250) — compile-verified, matrix operand order checked by reading, not run on hardware
//
#include <hip/hip_runtime.h>
#include <hip/hip_bf16.h>

typedef __attribute__((ext_vector_type(16))) _Float16 v16h;
typedef __attribute__((ext_vector_type(8)))  float    v8f;

#define HIDDEN 2048
#define NH 32
#define NKV 8
#define HD 64
#define BATCH 2
#define SEQ 2048
#define QKVD 3072   /* (NH + 2*NKV)*HD */

// ---------------------------------------------------------------------------
// WMMA fragment helpers (wave32, V_WMMA_F32_16X16X32_F16)
// A fragment (16x32 f16): lane L -> row = L%16, K-half = L/16.
//   VGPR i holds K pair: base(i) = (i<4) ? 2i : 16+2(i-4), K = base + half*8.
// B fragment (32x16 f16): mirrored, lane L -> col = L%16; stored in LDS as
//   Bt[n][k] (n-major) so the (k, k+1) pairs merge into b32/b128 LDS loads.
// ---------------------------------------------------------------------------
__device__ __forceinline__ v16h frag_a(const _Float16* p, int row0, int k0, int ld) {
  const int lane = threadIdx.x & 31;
  const int row  = row0 + (lane & 15);
  const int half = lane >> 4;
  v16h a;
#pragma unroll
  for (int i = 0; i < 8; ++i) {
    const int kb = ((i < 4) ? (2 * i) : (16 + 2 * (i - 4))) + half * 8;
    a[2 * i]     = p[row * ld + k0 + kb];
    a[2 * i + 1] = p[row * ld + k0 + kb + 1];
  }
  return a;
}

__device__ __forceinline__ v16h frag_b(const _Float16* p, int k0, int n0, int ld) {
  const int lane = threadIdx.x & 31;
  const int col  = n0 + (lane & 15);
  const int half = lane >> 4;
  v16h b;
#pragma unroll
  for (int i = 0; i < 8; ++i) {
    const int kb = ((i < 4) ? (2 * i) : (16 + 2 * (i - 4))) + half * 8;
    b[2 * i]     = p[col * ld + k0 + kb];
    b[2 * i + 1] = p[col * ld + k0 + kb + 1];
  }
  return b;
}

__device__ __forceinline__ v8f wmma_f16(v16h a, v16h b, v8f c) {
  return __builtin_amdgcn_wmma_f32_16x16x32_f16(false, a, false, b, (short)0, c,
                                                false, false);
}

// ---------------------------------------------------------------------------
// fp32 -> f16 conversion
// ---------------------------------------------------------------------------
__global__ void cvt_f32_f16(const float* __restrict__ in,
                            _Float16* __restrict__ out, int n) {
  int i = blockIdx.x * blockDim.x + threadIdx.x;
  if (i < n) out[i] = (_Float16)in[i];
}

// ---------------------------------------------------------------------------
// Generic GEMM: C[M,N](f32) = A[M,K](f16, row-major) * B[K,N](f16, row-major)
// Block tile 128x128, K-step 32, 256 threads = 8 waves (4 x 2 wave grid).
// Each wave owns a 32x64 output strip: 2 A-frags x 4 B-frags -> 8 WMMA
// per K-step from 6 fragment loads (2x better WMMA:LDS ratio than 16x32).
// ---------------------------------------------------------------------------
#define BM 128
#define BN 128
#define BK 32
#define GLD 40  /* BK + 8 halves pad: 80B row stride (16B aligned, bank-skewed) */

__global__ __launch_bounds__(256) void gemm_f16_f32(
    const _Float16* __restrict__ A, const _Float16* __restrict__ B,
    float* __restrict__ C, int M, int N, int K) {
  __shared__ __align__(16) _Float16 As[BM * GLD];   // 10 KB
  __shared__ __align__(16) _Float16 Bt[BN * GLD];   // 10 KB
  const int tid  = threadIdx.x;
  const int lane = tid & 31;
  const int wid  = tid >> 5;
  const int wm   = wid & 3;   // 0..3 -> 32-row strip
  const int wn   = wid >> 2;  // 0..1 -> 64-col strip
  const int m0   = blockIdx.y * BM;
  const int n0   = blockIdx.x * BN;

  v8f acc[2][4];
  v8f zf = {};
#pragma unroll
  for (int i = 0; i < 2; ++i)
#pragma unroll
    for (int f = 0; f < 4; ++f) acc[i][f] = zf;

  // A: 128 rows x 32 halves -> each thread 16 contiguous halves (2 x uint4)
  const int arow = tid >> 1;
  const int acol = (tid & 1) * 16;
  // B: 32 k-rows x 128 cols -> each thread 16 contiguous halves, scattered
  const int brow = tid >> 3;
  const int bcol = (tid & 7) * 16;

  for (int k0 = 0; k0 < K; k0 += BK) {
    __syncthreads();
    const _Float16* ag = A + (size_t)(m0 + arow) * K + k0 + acol;
    const _Float16* bg = B + (size_t)(k0 + brow) * N + n0 + bcol;
    {  // A tile
      *(uint4*)&As[arow * GLD + acol]     = *(const uint4*)ag;
      *(uint4*)&As[arow * GLD + acol + 8] = *(const uint4*)(ag + 8);
    }
    {  // B tile, stored transposed Bt[n][k]
      _Float16 tmp[16];
      *(uint4*)&tmp[0] = *(const uint4*)bg;
      *(uint4*)&tmp[8] = *(const uint4*)(bg + 8);
#pragma unroll
      for (int j = 0; j < 16; ++j) Bt[(bcol + j) * GLD + brow] = tmp[j];
    }
    if (k0 + BK < K) {  // prefetch next K tile -> global_prefetch_b8
      __builtin_prefetch((const void*)(ag + BK), 0, 0);
      __builtin_prefetch((const void*)(bg + (size_t)BK * N), 0, 0);
    }
    __syncthreads();

    v16h afr[2];
    afr[0] = frag_a(As, wm * 32, 0, GLD);
    afr[1] = frag_a(As, wm * 32 + 16, 0, GLD);
    v16h bfr[4];
#pragma unroll
    for (int f = 0; f < 4; ++f) bfr[f] = frag_b(Bt, 0, wn * 64 + f * 16, GLD);
#pragma unroll
    for (int i = 0; i < 2; ++i)
#pragma unroll
      for (int f = 0; f < 4; ++f) acc[i][f] = wmma_f16(afr[i], bfr[f], acc[i][f]);
  }

  // C/D layout: VGPR r -> row r (lanes 0-15) / r+8 (lanes 16-31), col = lane%16
#pragma unroll
  for (int i = 0; i < 2; ++i) {
    const int rbase = m0 + wm * 32 + i * 16 + ((lane >> 4) << 3);
    const int cbase = n0 + wn * 64 + (lane & 15);
#pragma unroll
    for (int f = 0; f < 4; ++f)
#pragma unroll
      for (int r = 0; r < 8; ++r)
        C[(size_t)(rbase + r) * N + cbase + f * 16] = acc[i][f][r];
  }
}

// ---------------------------------------------------------------------------
// RoPE (interleaved) + split qkv into head-major f16 buffers.
// qh: [B,NH,S,HD]  kh/vh: [B,NKV,S,HD]
// ---------------------------------------------------------------------------
__global__ void rope_split(const float* __restrict__ qkv,
                           _Float16* __restrict__ qh,
                           _Float16* __restrict__ kh,
                           _Float16* __restrict__ vh) {
  int idx = blockIdx.x * blockDim.x + threadIdx.x;
  const int total = BATCH * SEQ * (NH + 2 * NKV) * (HD / 2);
  if (idx >= total) return;
  const int j  = idx & 31;          // pair index 0..31
  int rest     = idx >> 5;
  const int hh = rest % (NH + 2 * NKV);
  rest        /= (NH + 2 * NKV);
  const int s  = rest % SEQ;
  const int b  = rest / SEQ;

  const float* row = qkv + (size_t)(b * SEQ + s) * QKVD;
  const float log_base = 9.210340371976184f;  // ln(10000)
  const float freq = __expf(-((float)(2 * j) / (float)HD) * log_base);
  float sn, cs;
  __sincosf((float)s * freq, &sn, &cs);

  if (hh < NH) {
    const float t1 = row[hh * HD + 2 * j];
    const float t2 = row[hh * HD + 2 * j + 1];
    _Float16* o = qh + ((size_t)(b * NH + hh) * SEQ + s) * HD;
    o[2 * j]     = (_Float16)(t1 * cs - t2 * sn);
    o[2 * j + 1] = (_Float16)(t1 * sn + t2 * cs);
  } else if (hh < NH + NKV) {
    const int kvh = hh - NH;
    const float t1 = row[HIDDEN + kvh * HD + 2 * j];
    const float t2 = row[HIDDEN + kvh * HD + 2 * j + 1];
    _Float16* o = kh + ((size_t)(b * NKV + kvh) * SEQ + s) * HD;
    o[2 * j]     = (_Float16)(t1 * cs - t2 * sn);
    o[2 * j + 1] = (_Float16)(t1 * sn + t2 * cs);
  } else {
    const int kvh = hh - NH - NKV;
    const float* v = row + HIDDEN + NKV * HD + kvh * HD;
    _Float16* o = vh + ((size_t)(b * NKV + kvh) * SEQ + s) * HD;
    o[2 * j]     = (_Float16)v[2 * j];
    o[2 * j + 1] = (_Float16)v[2 * j + 1];
  }
}

// ---------------------------------------------------------------------------
// Flash attention, causal, GQA (4 q-heads per kv-head).
// Block = 128 threads = 4 waves; block handles one 64-query tile of (b,h).
// Each wave owns a 16-query strip: QK^T (8 wmma) -> online softmax with
// 16-lane shfl_xor row reductions -> P via LDS (C-layout -> A-layout) ->
// P*V (8 wmma). Output written f16 into attn buffer [B*S, NH*HD].
// ---------------------------------------------------------------------------
#define ALD 72
__global__ __launch_bounds__(128) void flash_attn(
    const _Float16* __restrict__ qh, const _Float16* __restrict__ kh,
    const _Float16* __restrict__ vh, const int* __restrict__ mask,
    _Float16* __restrict__ attnh) {
  __shared__ __align__(16) _Float16 Qs[64 * ALD];   // [q][hd]
  __shared__ __align__(16) _Float16 Ks[64 * ALD];   // [key][hd]
  __shared__ __align__(16) _Float16 Vt[64 * ALD];   // [hd][key]
  __shared__ __align__(16) _Float16 Ps[64 * ALD];   // 4 waves x [16][64]

  const int tid   = threadIdx.x;
  const int lane  = tid & 31;
  const int wave  = tid >> 5;
  const int qt    = blockIdx.x;
  const int h     = blockIdx.y;
  const int b     = blockIdx.z;
  const int kvh   = h >> 2;        // groups = NH/NKV = 4
  const int qbase = qt * 64;
  const int rhalf = (lane >> 4) << 3;

  {  // load Q tile (64 x 64 halves)
    const int row = tid >> 1;
    const int col = (tid & 1) * 32;
    const _Float16* g =
        qh + ((size_t)(b * NH + h) * SEQ + (qbase + row)) * HD + col;
#pragma unroll
    for (int j = 0; j < 4; ++j)
      *(uint4*)&Qs[row * ALD + col + j * 8] = *(const uint4*)(g + j * 8);
  }

  float mrow[8], lrow[8];
  v8f oacc[4];
  v8f zf = {};
#pragma unroll
  for (int r = 0; r < 8; ++r) { mrow[r] = -1e30f; lrow[r] = 0.0f; }
#pragma unroll
  for (int f = 0; f < 4; ++f) oacc[f] = zf;

  const int nkt = qt + 1;  // causal: key tiles 0..qt
  for (int kt = 0; kt < nkt; ++kt) {
    __syncthreads();
    const _Float16* kg =
        kh + ((size_t)(b * NKV + kvh) * SEQ + (kt * 64 + (tid >> 1))) * HD +
        (tid & 1) * 32;
    const _Float16* vg =
        vh + ((size_t)(b * NKV + kvh) * SEQ + (kt * 64 + (tid >> 1))) * HD +
        (tid & 1) * 32;
    {  // K tile [key][hd]
      const int row = tid >> 1;
      const int col = (tid & 1) * 32;
#pragma unroll
      for (int j = 0; j < 4; ++j)
        *(uint4*)&Ks[row * ALD + col + j * 8] = *(const uint4*)(kg + j * 8);
    }
    {  // V tile transposed -> Vt[hd][key]
      const int key = tid >> 1;
      const int hd0 = (tid & 1) * 32;
      _Float16 tmp[32];
#pragma unroll
      for (int j = 0; j < 4; ++j)
        *(uint4*)&tmp[j * 8] = *(const uint4*)(vg + j * 8);
#pragma unroll
      for (int j = 0; j < 32; ++j) Vt[(hd0 + j) * ALD + key] = tmp[j];
    }
    if (kt + 1 < nkt) {  // prefetch next K/V tile -> global_prefetch_b8
      __builtin_prefetch((const void*)(kg + (size_t)64 * HD), 0, 0);
      __builtin_prefetch((const void*)(vg + (size_t)64 * HD), 0, 0);
    }
    __syncthreads();

    // ---- scores: S = Q * K^T  (16q x 64k per wave) ----
    v8f sc[4];
#pragma unroll
    for (int f = 0; f < 4; ++f) sc[f] = zf;
    v16h a0 = frag_a(Qs, wave * 16, 0, ALD);
    v16h a1 = frag_a(Qs, wave * 16, 32, ALD);
#pragma unroll
    for (int f = 0; f < 4; ++f) {
      v16h b0 = frag_b(Ks, 0, f * 16, ALD);
      v16h b1 = frag_b(Ks, 32, f * 16, ALD);
      sc[f] = wmma_f16(a0, b0, sc[f]);
      sc[f] = wmma_f16(a1, b1, sc[f]);
    }

    // ---- mask + scale + online softmax ----
    bool kok[4];
#pragma unroll
    for (int f = 0; f < 4; ++f) {
      const int key = kt * 64 + f * 16 + (lane & 15);
      kok[f] = mask[b * SEQ + key] > 0;
    }

    float pm[8];
#pragma unroll
    for (int r = 0; r < 8; ++r) {
      const int qrow = qbase + wave * 16 + r + rhalf;
      float best = -1e30f;
#pragma unroll
      for (int f = 0; f < 4; ++f) {
        const int key = kt * 64 + f * 16 + (lane & 15);
        float sv = sc[f][r] * 0.125f;  // 1/sqrt(HD)
        sv = (key <= qrow && kok[f]) ? sv : -1e9f;
        sc[f][r] = sv;
        best = fmaxf(best, sv);
      }
      pm[r] = best;
    }
#pragma unroll
    for (int r = 0; r < 8; ++r) {
#pragma unroll
      for (int off = 1; off < 16; off <<= 1)
        pm[r] = fmaxf(pm[r], __shfl_xor(pm[r], off, 32));
    }
#pragma unroll
    for (int r = 0; r < 8; ++r) {
      const float mnew = fmaxf(mrow[r], pm[r]);
      const float corr = __expf(mrow[r] - mnew);
      mrow[r] = mnew;
      float rs = 0.0f;
#pragma unroll
      for (int f = 0; f < 4; ++f) {
        const float p = __expf(sc[f][r] - mnew);
        rs += p;
        Ps[(wave * 16 + r + rhalf) * ALD + f * 16 + (lane & 15)] = (_Float16)p;
      }
#pragma unroll
      for (int off = 1; off < 16; off <<= 1) rs += __shfl_xor(rs, off, 32);
      lrow[r] = lrow[r] * corr + rs;
#pragma unroll
      for (int f = 0; f < 4; ++f) oacc[f][r] *= corr;
    }
    __syncthreads();  // P visible (C-layout -> A-layout via LDS)

    // ---- O += P * V ----
    v16h p0 = frag_a(Ps, wave * 16, 0, ALD);
    v16h p1 = frag_a(Ps, wave * 16, 32, ALD);
#pragma unroll
    for (int f = 0; f < 4; ++f) {
      v16h vb0 = frag_b(Vt, 0, f * 16, ALD);
      v16h vb1 = frag_b(Vt, 32, f * 16, ALD);
      oacc[f] = wmma_f16(p0, vb0, oacc[f]);
      oacc[f] = wmma_f16(p1, vb1, oacc[f]);
    }
  }

  // ---- epilogue: normalize, query-mask, write f16 attn [B*S, NH*HD] ----
#pragma unroll
  for (int r = 0; r < 8; ++r) {
    const int qrow = qbase + wave * 16 + r + rhalf;
    const float inv = (lrow[r] > 0.0f) ? (1.0f / lrow[r]) : 0.0f;
    const float qm = (mask[b * SEQ + qrow] > 0) ? 1.0f : 0.0f;
#pragma unroll
    for (int f = 0; f < 4; ++f) {
      const float val = oacc[f][r] * inv * qm;
      attnh[((size_t)(b * SEQ) + qrow) * (NH * HD) + h * HD + f * 16 +
            (lane & 15)] = (_Float16)val;
    }
  }
}

// ---------------------------------------------------------------------------
// Launch
// ---------------------------------------------------------------------------
extern "C" void kernel_launch(void* const* d_in, const int* in_sizes, int n_in,
                              void* d_out, int out_size, void* d_ws,
                              size_t ws_size, hipStream_t stream) {
  const float* x    = (const float*)d_in[0];
  const int*   mask = (const int*)d_in[1];
  const float* wqkv = (const float*)d_in[2];
  const float* wo   = (const float*)d_in[3];
  float* out = (float*)d_out;

  char* ws = (char*)d_ws;
  size_t off = 0;
  auto alloc = [&](size_t bytes) -> void* {
    void* p = ws + off;
    off += (bytes + 255) & ~(size_t)255;
    return p;
  };
  _Float16* xh    = (_Float16*)alloc((size_t)BATCH * SEQ * HIDDEN * 2);
  _Float16* wqkvh = (_Float16*)alloc((size_t)HIDDEN * QKVD * 2);
  _Float16* woh   = (_Float16*)alloc((size_t)HIDDEN * HIDDEN * 2);
  float*    qkvf  = (float*)alloc((size_t)BATCH * SEQ * QKVD * 4);
  _Float16* qhb   = (_Float16*)alloc((size_t)BATCH * NH * SEQ * HD * 2);
  _Float16* khb   = (_Float16*)alloc((size_t)BATCH * NKV * SEQ * HD * 2);
  _Float16* vhb   = (_Float16*)alloc((size_t)BATCH * NKV * SEQ * HD * 2);
  _Float16* attnh = (_Float16*)alloc((size_t)BATCH * SEQ * HIDDEN * 2);

  const int n1 = BATCH * SEQ * HIDDEN;
  cvt_f32_f16<<<(n1 + 255) / 256, 256, 0, stream>>>(x, xh, n1);
  const int n2 = HIDDEN * QKVD;
  cvt_f32_f16<<<(n2 + 255) / 256, 256, 0, stream>>>(wqkv, wqkvh, n2);
  const int n3 = HIDDEN * HIDDEN;
  cvt_f32_f16<<<(n3 + 255) / 256, 256, 0, stream>>>(wo, woh, n3);

  dim3 g1(QKVD / BN, (BATCH * SEQ) / BM);
  gemm_f16_f32<<<g1, 256, 0, stream>>>(xh, wqkvh, qkvf, BATCH * SEQ, QKVD,
                                       HIDDEN);

  const int nr = BATCH * SEQ * (NH + 2 * NKV) * (HD / 2);
  rope_split<<<(nr + 255) / 256, 256, 0, stream>>>(qkvf, qhb, khb, vhb);

  dim3 g2(SEQ / 64, NH, BATCH);
  flash_attn<<<g2, 128, 0, stream>>>(qhb, khb, vhb, mask, attnh);

  dim3 g3(HIDDEN / BN, (BATCH * SEQ) / BM);
  gemm_f16_f32<<<g3, 256, 0, stream>>>(attnh, woh, out, BATCH * SEQ, HIDDEN,
                                       HIDDEN);
}
